// Sequence_13245679141348
// MI455X (gfx1250) — matrix operand order repaired and checked by reading, not compiled, not run on hardware
//
#include <hip/hip_runtime.h>
#include <hip/hip_bf16.h>

typedef _Float16 v16h __attribute__((ext_vector_type(16)));
typedef _Float16 v8h  __attribute__((ext_vector_type(8)));
typedef float    v8f  __attribute__((ext_vector_type(8)));

#define LSTM_T  2048
#define LSTM_B  64
#define LSTM_F  128
#define LSTM_H  256
#define LSTM_G  1024                 // 4*H
#define LSTM_M  (LSTM_T * LSTM_B)   // 131072 rows (t*B + b)
#define LSTM_MT (LSTM_M / 16)       // 8192 M-tiles

// ---------------------------------------------------------------------------
// WMMA fragment helpers (f16 16x16x32, layouts per CDNA5 ISA 7.12.2)
// ---------------------------------------------------------------------------
__device__ __forceinline__ v16h make_frag(v8h lo, v8h hi) {
  v16h r;
#pragma unroll
  for (int j = 0; j < 8; ++j) { r[j] = lo[j]; r[j + 8] = hi[j]; }
  return r;
}

// A fragment (16x32) from row-major f16 src (leading dim ld), tile at (row0,k0).
// lane l (l<16): row=l, K = {k0..k0+7, k0+16..k0+23}; lanes 16..31: +8 K offset.
__device__ __forceinline__ v16h load_a_frag(const _Float16* __restrict__ src,
                                            int ld, int row0, int k0, int lane) {
  const int row = row0 + (lane & 15);
  const int kb  = k0 + ((lane >> 4) << 3);
  const _Float16* p = src + (size_t)row * ld + kb;
  v8h lo = *(const v8h*)p;
  v8h hi = *(const v8h*)(p + 16);
  return make_frag(lo, hi);
}

// B fragment (32x16) where B[k][n] = W[n][k], W row-major (rows=N, ld=K).
// lane l: col = l&15, K = k0 + 8*(l>>4) + {0..7, 16..23} (matches A fragment).
__device__ __forceinline__ v16h load_b_frag(const _Float16* __restrict__ W,
                                            int ld, int n0, int k0, int lane) {
  const int n  = n0 + (lane & 15);
  const int kb = k0 + ((lane >> 4) << 3);
  const _Float16* p = W + (size_t)n * ld + kb;
  v8h lo = *(const v8h*)p;
  v8h hi = *(const v8h*)(p + 16);
  return make_frag(lo, hi);
}

__device__ __forceinline__ float sigmoidf_dev(float x) {
  return 1.0f / (1.0f + __expf(-x));
}

// ---------------------------------------------------------------------------
// fp32 -> fp16 conversion
// ---------------------------------------------------------------------------
__global__ void f32_to_f16_kernel(const float* __restrict__ src,
                                  _Float16* __restrict__ dst, size_t n) {
  size_t i = (size_t)blockIdx.x * blockDim.x + threadIdx.x;
  if (i < n) dst[i] = (_Float16)src[i];
}

// ---------------------------------------------------------------------------
// Time-parallel input projection: P = X @ Wih^T + (bih + bhh)
// X: (M, K) f16 row-major; W: (1024, K) f16 row-major; P stored tile-serialized
// f16: P[mtile][nt][lane][v] (raw C-fragment layout, 256 halves per tile).
// block = 256 threads (8 waves); each wave computes one 16x16 tile.
// grid.x = MT * 8 (8 n-tile groups of 8 per m-tile).
// ---------------------------------------------------------------------------
__global__ __launch_bounds__(256) void xproj_gemm_kernel(
    const _Float16* __restrict__ X, const _Float16* __restrict__ W,
    const float* __restrict__ bih, const float* __restrict__ bhh,
    _Float16* __restrict__ P, int K) {
  const int lane  = threadIdx.x & 31;
  const int wv    = threadIdx.x >> 5;
  const int mtile = blockIdx.x >> 3;
  const int nt    = ((blockIdx.x & 7) << 3) + wv;

  v8f acc = {};
  for (int k0 = 0; k0 < K; k0 += 32) {
    v16h a = load_a_frag(X, K, mtile * 16, k0, lane);
    v16h b = load_b_frag(W, K, nt * 16, k0, lane);
    acc = __builtin_amdgcn_wmma_f32_16x16x32_f16(false, a, false, b,
                                                 (short)0, acc, false, false);
  }
  const int n = nt * 16 + (lane & 15);
  const float bsum = bih[n] + bhh[n];
  v8h outv;
#pragma unroll
  for (int v = 0; v < 8; ++v) outv[v] = (_Float16)(acc[v] + bsum);
  *(v8h*)(P + ((size_t)mtile * 64 + nt) * 256 + lane * 8) = outv;
}

// ---------------------------------------------------------------------------
// Persistent sequential LSTM scan for one layer. Single workgroup, 32 waves.
//   gates_t = P[t] + h_{t-1} @ Whh^T ; c,h update; h -> LDS + global.
// Wave w: m-tile = w&3 (batch 16-rows), h-tiles {2*(w>>2), 2*(w>>2)+1};
// for each h-tile it owns all four gate tiles (i,f,g,o), so the cell update
// is wave-local. c lives in accumulator registers across all T steps.
// ---------------------------------------------------------------------------
__global__ __launch_bounds__(1024) void lstm_scan_kernel(
    const _Float16* __restrict__ P,    // [T*4 m-tiles][64 n-tiles][32][8] f16
    const _Float16* __restrict__ Whh,  // (1024, 256) f16 row-major
    _Float16* __restrict__ h_out,      // (T*64, 256) f16 row-major
    int T) {
  __shared__ _Float16 h_lds[LSTM_B][LSTM_H];  // 32 KB

  const int tid  = threadIdx.x;
  const int lane = tid & 31;
  const int wv   = tid >> 5;
  const int mt   = wv & 3;
  const int htb  = (wv >> 2) << 1;  // 0,2,...,14

  // zero initial hidden state
  for (int i = tid; i < LSTM_B * LSTM_H; i += 1024)
    ((_Float16*)h_lds)[i] = (_Float16)0.0f;
  __syncthreads();

  v8f cfr[2];
  cfr[0] = (v8f){};
  cfr[1] = (v8f){};

  for (int t = 0; t < T; ++t) {
    const _Float16* Pt = P + ((size_t)(t * 4 + mt) * 64) * 256;

    // init gate accumulators from precomputed x-projection (f16 -> f32)
    v8f acc[8];
#pragma unroll
    for (int q = 0; q < 2; ++q) {
      const int ht = htb + q;
#pragma unroll
      for (int g = 0; g < 4; ++g) {
        const int nt = g * 16 + ht;
        v8h pv = *(const v8h*)(Pt + (size_t)nt * 256 + lane * 8);
        v8f a;
#pragma unroll
        for (int j = 0; j < 8; ++j) a[j] = (float)pv[j];
        acc[q * 4 + g] = a;
      }
    }

    // prefetch next step's x-projection tiles while we do the GEMM
    if (t + 1 < T) {
      const _Float16* Pn =
          P + ((size_t)((t + 1) * 4 + mt) * 64 + htb * 4) * 256 + lane * 8;
      __builtin_prefetch(Pn, 0, 0);
    }

    // gates += h_{t-1} @ Whh^T   (K = 256, 8 chunks of 32)
#pragma unroll
    for (int kc = 0; kc < 8; ++kc) {
      const int k0  = kc * 32;
      const int row = mt * 16 + (lane & 15);
      const int kb  = k0 + ((lane >> 4) << 3);
      v8h alo = *(const v8h*)&h_lds[row][kb];
      v8h ahi = *(const v8h*)&h_lds[row][kb + 16];
      v16h afr = make_frag(alo, ahi);
#pragma unroll
      for (int q = 0; q < 2; ++q) {
        const int ht = htb + q;
#pragma unroll
        for (int g = 0; g < 4; ++g) {
          const int nt = g * 16 + ht;
          v16h bfr = load_b_frag(Whh, LSTM_H, nt * 16, k0, lane);
          acc[q * 4 + g] = __builtin_amdgcn_wmma_f32_16x16x32_f16(
              false, afr, false, bfr, (short)0, acc[q * 4 + g], false, false);
        }
      }
    }

    __syncthreads();  // all waves done reading h_lds for this step

    // activations + cell/hidden update (wave-local; c stays in registers)
#pragma unroll
    for (int q = 0; q < 2; ++q) {
      const int ht = htb + q;
      v8f ig = acc[q * 4 + 0];
      v8f fg = acc[q * 4 + 1];
      v8f gg = acc[q * 4 + 2];
      v8f og = acc[q * 4 + 3];
#pragma unroll
      for (int v = 0; v < 8; ++v) {
        const float iv = sigmoidf_dev(ig[v]);
        const float fv = sigmoidf_dev(fg[v]);
        const float gv = tanhf(gg[v]);
        const float ov = sigmoidf_dev(og[v]);
        const float cv = fv * cfr[q][v] + iv * gv;
        cfr[q][v] = cv;
        const float hv = ov * tanhf(cv);
        const int b  = mt * 16 + v + ((lane >> 4) << 3);
        const int hh = ht * 16 + (lane & 15);
        const _Float16 hvh = (_Float16)hv;
        h_lds[b][hh] = hvh;
        h_out[((size_t)t * LSTM_B + b) * LSTM_H + hh] = hvh;
      }
    }

    __syncthreads();  // h writes visible before next step's GEMM
  }
}

// ---------------------------------------------------------------------------
// Final linear projection: out[m] = h[m,:] . Wl + bl
// ---------------------------------------------------------------------------
__global__ void final_proj_kernel(const _Float16* __restrict__ h,
                                  const float* __restrict__ Wl,
                                  const float* __restrict__ bl,
                                  float* __restrict__ out, int M) {
  const int m = blockIdx.x * blockDim.x + threadIdx.x;
  if (m >= M) return;
  const _Float16* hp = h + (size_t)m * LSTM_H;
  float s = bl[0];
  for (int k = 0; k < LSTM_H; k += 8) {
    v8h hv = *(const v8h*)(hp + k);
#pragma unroll
    for (int j = 0; j < 8; ++j) s += (float)hv[j] * Wl[k + j];
  }
  out[m] = s;
}

// ---------------------------------------------------------------------------
extern "C" void kernel_launch(void* const* d_in, const int* in_sizes, int n_in,
                              void* d_out, int out_size, void* d_ws,
                              size_t ws_size, hipStream_t stream) {
  (void)in_sizes; (void)n_in; (void)out_size; (void)ws_size;

  const float* x    = (const float*)d_in[0];
  const float* Wih0 = (const float*)d_in[1];
  const float* Whh0 = (const float*)d_in[2];
  const float* bih0 = (const float*)d_in[3];
  const float* bhh0 = (const float*)d_in[4];
  const float* Wih1 = (const float*)d_in[5];
  const float* Whh1 = (const float*)d_in[6];
  const float* bih1 = (const float*)d_in[7];
  const float* bhh1 = (const float*)d_in[8];
  const float* Wih2 = (const float*)d_in[9];
  const float* Whh2 = (const float*)d_in[10];
  const float* bih2 = (const float*)d_in[11];
  const float* bhh2 = (const float*)d_in[12];
  const float* Wl   = (const float*)d_in[13];
  const float* bl   = (const float*)d_in[14];
  float* out = (float*)d_out;

  char* ws = (char*)d_ws;
  size_t off = 0;
  auto take = [&](size_t halves) -> _Float16* {
    _Float16* p = (_Float16*)(ws + off);
    off += halves * sizeof(_Float16);
    off = (off + 255) & ~(size_t)255;
    return p;
  };
  _Float16* x16   = take((size_t)LSTM_M * LSTM_F);   //  32 MB
  _Float16* wih0h = take((size_t)LSTM_G * LSTM_F);
  _Float16* whh0h = take((size_t)LSTM_G * LSTM_H);
  _Float16* wih1h = take((size_t)LSTM_G * LSTM_H);
  _Float16* whh1h = take((size_t)LSTM_G * LSTM_H);
  _Float16* wih2h = take((size_t)LSTM_G * LSTM_H);
  _Float16* whh2h = take((size_t)LSTM_G * LSTM_H);
  _Float16* Pbuf  = take((size_t)LSTM_MT * 64 * 256); // 256 MB xproj tiles
  _Float16* hA    = take((size_t)LSTM_M * LSTM_H);    //  64 MB
  _Float16* hB    = take((size_t)LSTM_M * LSTM_H);    //  64 MB

  auto cvt = [&](const float* s, _Float16* d, size_t n) {
    f32_to_f16_kernel<<<(unsigned)((n + 255) / 256), 256, 0, stream>>>(s, d, n);
  };
  cvt(x,    x16,   (size_t)LSTM_M * LSTM_F);
  cvt(Wih0, wih0h, (size_t)LSTM_G * LSTM_F);
  cvt(Whh0, whh0h, (size_t)LSTM_G * LSTM_H);
  cvt(Wih1, wih1h, (size_t)LSTM_G * LSTM_H);
  cvt(Whh1, whh1h, (size_t)LSTM_G * LSTM_H);
  cvt(Wih2, wih2h, (size_t)LSTM_G * LSTM_H);
  cvt(Whh2, whh2h, (size_t)LSTM_G * LSTM_H);

  const unsigned gemm_grid = LSTM_MT * 8;

  // layer 0
  xproj_gemm_kernel<<<gemm_grid, 256, 0, stream>>>(x16, wih0h, bih0, bhh0,
                                                   Pbuf, LSTM_F);
  lstm_scan_kernel<<<1, 1024, 0, stream>>>(Pbuf, whh0h, hA, LSTM_T);
  // layer 1
  xproj_gemm_kernel<<<gemm_grid, 256, 0, stream>>>(hA, wih1h, bih1, bhh1,
                                                   Pbuf, LSTM_H);
  lstm_scan_kernel<<<1, 1024, 0, stream>>>(Pbuf, whh1h, hB, LSTM_T);
  // layer 2
  xproj_gemm_kernel<<<gemm_grid, 256, 0, stream>>>(hB, wih2h, bih2, bhh2,
                                                   Pbuf, LSTM_H);
  lstm_scan_kernel<<<1, 1024, 0, stream>>>(Pbuf, whh2h, hA, LSTM_T);
  // output projection
  final_proj_kernel<<<(LSTM_M + 255) / 256, 256, 0, stream>>>(hA, Wl, bl, out,
                                                              LSTM_M);
}
